// GNNTower_19396072308944
// MI455X (gfx1250) — matrix operand order repaired and checked by reading, not compiled, not run on hardware
//
#include <hip/hip_runtime.h>
#include <hip/hip_bf16.h>

// ---------------------------------------------------------------------------
// CDNA5 (gfx1250) WMMA GNN tower.
// Dense GEMMs run on v_wmma_f32_16x16x32_bf16 (bf16 in, f32 accumulate).
// Weights are pre-converted once to bf16, transposed (W^T[n][K]), so B
// fragments load straight from global/L2; LDS is used only for the A tile.
// ---------------------------------------------------------------------------

typedef __attribute__((ext_vector_type(16))) __bf16 v16bf;
typedef __attribute__((ext_vector_type(8)))  float  v8f;

#define HX 128
#define A_PITCH 72    // 64 K-elements + 8 pad (bf16); row stride 144B (16B mult)
#define BLK_ROWS 64   // rows per block (4 row-tiles of 16)

// --- WMMA fragment loaders in contiguous k order (k = 16*(lane/16) + elem), same for A and B --

// 16-bit A matrix 16x32 (MxK), from LDS:
//   lanes 0-15 : row M=lane,    elems 0..15 -> K=0..15
//   lanes16-31 : row M=lane-16, elems 0..15 -> K=16..31
__device__ inline v16bf load_frag_a(const __bf16* smem, int m0, int k0) {
  const int lane = threadIdx.x & 31;
  const int m    = m0 + (lane & 15);
  const int koff = (lane >> 4) * 16;
  const __bf16* p = smem + m * A_PITCH + k0 + koff;
  v16bf a;
#pragma unroll
  for (int i = 0; i < 8; ++i) a[i] = p[i];          // contiguous 16B
#pragma unroll
  for (int i = 0; i < 8; ++i) a[8 + i] = p[8 + i]; // contiguous 16B
  return a;
}

// 16-bit B matrix 32x16 (KxN), direct from global bf16 W^T[n][K]:
//   lanes 0-15: col N=lane, K=ks..ks+15 ; lanes 16-31: col N=lane-16, K=+16
__device__ inline v16bf load_frag_b_g(const __bf16* __restrict__ wt,
                                      int ldwt, int n0, int ks) {
  const int lane = threadIdx.x & 31;
  const __bf16* p = wt + (long)(n0 + (lane & 15)) * ldwt + ks + (lane >> 4) * 16;
  v16bf b;
#pragma unroll
  for (int i = 0; i < 16; ++i) b[i] = p[i];         // contiguous 32B
  return b;
}

// ---------------------------------------------------------------------------
// Generic fused dense layer:  out[M, ncols] = act( concat(segs) @ W + bias )
// Segment with ld==0 is a broadcast row (time embedding). All segment K sizes
// are multiples of 128 and K steps are 64 wide / 64-aligned, so each step lies
// entirely in one segment -> segment selection is uniform (SGPR) per step.
// Block: 64 rows x 128 cols, 8 waves, 4 accumulator tiles per wave,
// 8 WMMAs per wave per staging barrier (64-deep K step).
// ---------------------------------------------------------------------------
struct DenseArgs {
  const float*  seg_ptr[5];
  int           seg_k[5];
  int           seg_ld[5];   // 0 => broadcast row
  int           nseg;
  int           M;
  int           Ktot;        // multiple of 64
  const __bf16* Wt;          // bf16 W^T : [ncols_total][Ktot]
  int           ldwt;        // == Ktot
  const float*  bias;        // indexed by absolute output column
  float*        out;
  int           ldo;
  int           relu;
};

__global__ __launch_bounds__(256) void dense_wmma_kernel(DenseArgs a) {
  __shared__ __bf16 sA[BLK_ROWS * A_PITCH];   // A step [64 rows][64 k]

  const int tid  = threadIdx.x;
  const int lane = tid & 31;
  const int wid  = tid >> 5;
  const int row0 = blockIdx.x * BLK_ROWS;
  const int n0   = blockIdx.y * 128;

  const int rt     = wid & 3;          // row tile 0..3 (16 rows each)
  const int ntBase = (wid >> 2) * 4;   // 4 of 8 n-tiles per wave
  const int nn0 = n0 + ntBase * 16;

  const bool fullTile = (row0 + BLK_ROWS) <= a.M;

  v8f acc[4] = {v8f{}, v8f{}, v8f{}, v8f{}};

  for (int ks = 0; ks < a.Ktot; ks += 64) {
    // ---- uniform (scalar) segment selection for this whole K step ----
    int s = 0, base = 0;
    while (s < a.nseg - 1 && ks >= base + a.seg_k[s]) { base += a.seg_k[s]; ++s; }
    const float* segp  = a.seg_ptr[s];
    const int    segld = a.seg_ld[s];
    const int    lk0   = ks - base;

    // ---- stage A step: float2 load -> 2x cvt -> ds_store_b32 ----
    if (fullTile) {
#pragma unroll
      for (int it = 0; it < 8; ++it) {
        const int idx = it * 256 + tid;           // 64 rows * 32 float2
        const int r   = idx >> 5;
        const int kp  = (idx & 31) * 2;
        const float* p = segld ? (segp + (long)(row0 + r) * segld + lk0 + kp)
                               : (segp + lk0 + kp);
        const float2 v = *(const float2*)p;
        __bf16* d = &sA[r * A_PITCH + kp];
        d[0] = (__bf16)v.x;
        d[1] = (__bf16)v.y;
      }
    } else {
#pragma unroll
      for (int it = 0; it < 8; ++it) {
        const int idx = it * 256 + tid;
        const int r   = idx >> 5;
        const int kp  = (idx & 31) * 2;
        float2 v = make_float2(0.0f, 0.0f);
        if (row0 + r < a.M) {
          const float* p = segld ? (segp + (long)(row0 + r) * segld + lk0 + kp)
                                 : (segp + lk0 + kp);
          v = *(const float2*)p;
        }
        __bf16* d = &sA[r * A_PITCH + kp];
        d[0] = (__bf16)v.x;
        d[1] = (__bf16)v.y;
      }
    }
    __syncthreads();

    // ---- two 32-deep halves; B loads grouped ahead of the 4 WMMAs ----
#pragma unroll
    for (int h = 0; h < 2; ++h) {
      const int kh = ks + h * 32;
      const v16bf af = load_frag_a(sA, rt * 16, h * 32);
      const v16bf b0 = load_frag_b_g(a.Wt, a.ldwt, nn0,      kh);
      const v16bf b1 = load_frag_b_g(a.Wt, a.ldwt, nn0 + 16, kh);
      const v16bf b2 = load_frag_b_g(a.Wt, a.ldwt, nn0 + 32, kh);
      const v16bf b3 = load_frag_b_g(a.Wt, a.ldwt, nn0 + 48, kh);
      acc[0] = __builtin_amdgcn_wmma_f32_16x16x32_bf16(false, af, false, b0,
                                                       (short)0, acc[0], false, false);
      acc[1] = __builtin_amdgcn_wmma_f32_16x16x32_bf16(false, af, false, b1,
                                                       (short)0, acc[1], false, false);
      acc[2] = __builtin_amdgcn_wmma_f32_16x16x32_bf16(false, af, false, b2,
                                                       (short)0, acc[2], false, false);
      acc[3] = __builtin_amdgcn_wmma_f32_16x16x32_bf16(false, af, false, b3,
                                                       (short)0, acc[3], false, false);
    }
    __syncthreads();
  }

  // ---- epilogue: bias + activation + f32 store ----
  // C/D layout: VGPR r -> row (r + 8*(lane>>4)), col (lane&15) within tile.
  const int mLocal = rt * 16 + (lane >> 4) * 8;
  const int nc     = lane & 15;
#pragma unroll
  for (int t = 0; t < 4; ++t) {
    const int col = nn0 + t * 16 + nc;
    const float bv = a.bias[col];
#pragma unroll
    for (int r = 0; r < 8; ++r) {
      const int row = row0 + mLocal + r;
      if (row < a.M) {
        float v = acc[t][r] + bv;
        if (a.relu) v = fmaxf(v, 0.0f);
        a.out[(long)row * a.ldo + col] = v;
      }
    }
  }
}

// ---------------------------------------------------------------------------
// One-time weight conversion: W f32 [K][Nc] -> W^T bf16 [Nc][K]
// ---------------------------------------------------------------------------
__global__ __launch_bounds__(256) void wt_convert_kernel(
    const float* __restrict__ W, __bf16* __restrict__ Wt, int K, int Nc) {
  const long total = (long)K * Nc;
  for (long i = blockIdx.x * 256L + threadIdx.x; i < total;
       i += (long)gridDim.x * 256L) {
    const int n = (int)(i / K);
    const int k = (int)(i - (long)n * K);
    Wt[i] = (__bf16)W[(long)k * Nc + n];   // writes coalesced over k
  }
}

// ---------------------------------------------------------------------------
// Sparse aggregation: aggr[row[e]] += val[e] * h[col[e]]   (one wave per edge)
// ---------------------------------------------------------------------------
__global__ __launch_bounds__(256) void edge_aggregate_kernel(
    const int* __restrict__ erow, const int* __restrict__ ecol,
    const float* __restrict__ eval, const float* __restrict__ h,
    float* __restrict__ aggr, int E) {
  const int lane   = threadIdx.x & 31;
  const int wave   = (blockIdx.x * blockDim.x + threadIdx.x) >> 5;
  const int nwaves = (gridDim.x * blockDim.x) >> 5;
  for (int e = wave; e < E; e += nwaves) {
    const int c = ecol[e];
    const int r = erow[e];
    const float v = eval[e];
    const float4 m = ((const float4*)(h + (long)c * HX))[lane];
    float* dst = aggr + (long)r * HX + lane * 4;
    unsafeAtomicAdd(dst + 0, v * m.x);
    unsafeAtomicAdd(dst + 1, v * m.y);
    unsafeAtomicAdd(dst + 2, v * m.z);
    unsafeAtomicAdd(dst + 3, v * m.w);
  }
}

// ---------------------------------------------------------------------------
// LayerNorm over 128 features; one wave per row, shfl_xor wave32 reduction.
// ---------------------------------------------------------------------------
__global__ __launch_bounds__(256) void layernorm_kernel(
    const float* __restrict__ x, const float* __restrict__ g,
    const float* __restrict__ b, float* __restrict__ y, int M) {
  const int lane   = threadIdx.x & 31;
  const int wave   = (blockIdx.x * blockDim.x + threadIdx.x) >> 5;
  const int nwaves = (gridDim.x * blockDim.x) >> 5;
  const float4 gg = ((const float4*)g)[lane];
  const float4 bb = ((const float4*)b)[lane];
  for (int r = wave; r < M; r += nwaves) {
    const float4 v = ((const float4*)(x + (long)r * HX))[lane];
    float s  = v.x + v.y + v.z + v.w;
    float ss = v.x * v.x + v.y * v.y + v.z * v.z + v.w * v.w;
#pragma unroll
    for (int o = 16; o > 0; o >>= 1) {
      s  += __shfl_xor(s, o, 32);
      ss += __shfl_xor(ss, o, 32);
    }
    const float mu  = s * (1.0f / 128.0f);
    const float var = ss * (1.0f / 128.0f) - mu * mu;
    const float rs  = rsqrtf(var + 1e-5f);
    float4 o4;
    o4.x = (v.x - mu) * rs * gg.x + bb.x;
    o4.y = (v.y - mu) * rs * gg.y + bb.y;
    o4.z = (v.z - mu) * rs * gg.z + bb.z;
    o4.w = (v.w - mu) * rs * gg.w + bb.w;
    ((float4*)(y + (long)r * HX))[lane] = o4;
  }
}

// ---------------------------------------------------------------------------
// Time embedding MLP: h_t = relu(relu(t*Wt1 + bt1) @ Wt2 + bt2)   (1 block)
// ---------------------------------------------------------------------------
__global__ __launch_bounds__(128) void time_mlp_kernel(
    const float* __restrict__ t, const float* __restrict__ Wt1,
    const float* __restrict__ bt1, const float* __restrict__ Wt2,
    const float* __restrict__ bt2, float* __restrict__ h_t) {
  __shared__ float s1[128];
  const int j = threadIdx.x;
  s1[j] = fmaxf(t[0] * Wt1[j] + bt1[j], 0.0f);
  __syncthreads();
  float acc = bt2[j];
  for (int i = 0; i < 128; ++i) acc += s1[i] * Wt2[i * 128 + j];
  h_t[j] = fmaxf(acc, 0.0f);
}

// ---------------------------------------------------------------------------
// Host-side launch
// ---------------------------------------------------------------------------
static inline DenseArgs make_dense1(const float* A, int ld, int K, int M,
                                    const __bf16* Wt, const float* bias,
                                    float* out, int ldo, int relu) {
  DenseArgs a = {};
  a.seg_ptr[0] = A; a.seg_k[0] = K; a.seg_ld[0] = ld;
  a.nseg = 1; a.M = M; a.Ktot = K;
  a.Wt = Wt; a.ldwt = K; a.bias = bias; a.out = out; a.ldo = ldo; a.relu = relu;
  return a;
}

extern "C" void kernel_launch(void* const* d_in, const int* in_sizes, int n_in,
                              void* d_out, int out_size, void* d_ws, size_t ws_size,
                              hipStream_t stream) {
  const float* t_float = (const float*)d_in[0];
  const float* X       = (const float*)d_in[1];
  const int*   erow    = (const int*)  d_in[2];
  const int*   ecol    = (const int*)  d_in[3];
  const float* eval    = (const float*)d_in[4];
  const float* Wt1 = (const float*)d_in[5],  *bt1 = (const float*)d_in[6];
  const float* Wt2 = (const float*)d_in[7],  *bt2 = (const float*)d_in[8];
  const float* Wx1 = (const float*)d_in[9],  *bx1 = (const float*)d_in[10];
  const float* Wx2 = (const float*)d_in[11], *bx2 = (const float*)d_in[12];
  const float* gnnW = (const float*)d_in[13], *gnnB = (const float*)d_in[14];
  const float* lnG  = (const float*)d_in[15], *lnB  = (const float*)d_in[16];
  const float* Wo1 = (const float*)d_in[17], *bo1 = (const float*)d_in[18];
  const float* Wo2 = (const float*)d_in[19], *bo2 = (const float*)d_in[20];
  float* out = (float*)d_out;

  const int N = in_sizes[1] / 128;
  const int E = in_sizes[2];

  // Workspace layout (floats, then bf16 weights):
  //   h[0..3] : 4 * N*128
  //   h_t     : 256 (padded)
  //   aggr    : N*128
  //   tmp     : N*640
  //   bf16 W^T: Wx1,Wx2 (128*128 each), gnn (3*128*256), Wo1 (640*640),
  //             Wo2 (128*640)
  float* ws   = (float*)d_ws;
  float* h0   = ws;
  float* h_t  = ws + 4L * N * 128;
  float* aggr = h_t + 256;
  float* tmp  = aggr + (long)N * 128;
  __bf16* wtX1 = (__bf16*)(tmp + 640L * N);
  __bf16* wtX2 = wtX1 + 128 * 128;
  __bf16* wtG  = wtX2 + 128 * 128;        // 3 layers, each [128][256]
  __bf16* wtO1 = wtG + 3L * 128 * 256;    // [640][640]
  __bf16* wtO2 = wtO1 + 640L * 640;       // [128][640]

  const dim3 blk(256);
  const dim3 grid1((N + BLK_ROWS - 1) / BLK_ROWS, 1);

  // 0) one-time weight conversion (f32 [K][Nc] -> bf16 W^T [Nc][K])
  wt_convert_kernel<<<64, blk, 0, stream>>>(Wx1, wtX1, 128, 128);
  wt_convert_kernel<<<64, blk, 0, stream>>>(Wx2, wtX2, 128, 128);
  for (int i = 0; i < 3; ++i)
    wt_convert_kernel<<<128, blk, 0, stream>>>(gnnW + (long)i * 256 * 128,
                                               wtG + (long)i * 128 * 256,
                                               256, 128);
  wt_convert_kernel<<<1024, blk, 0, stream>>>(Wo1, wtO1, 640, 640);
  wt_convert_kernel<<<256, blk, 0, stream>>>(Wo2, wtO2, 640, 128);

  // 1) time embedding
  time_mlp_kernel<<<1, 128, 0, stream>>>(t_float, Wt1, bt1, Wt2, bt2, h_t);

  // 2) input MLP: tmp = relu(X@Wx1+bx1); h0 = relu(tmp@Wx2+bx2)
  {
    DenseArgs a = make_dense1(X, 128, 128, N, wtX1, bx1, tmp, 128, 1);
    dense_wmma_kernel<<<grid1, blk, 0, stream>>>(a);
  }
  {
    DenseArgs a = make_dense1(tmp, 128, 128, N, wtX2, bx2, h0, 128, 1);
    dense_wmma_kernel<<<grid1, blk, 0, stream>>>(a);
  }

  // 3) GNN layers
  for (int i = 0; i < 3; ++i) {
    const float* h_in = h0 + (long)i * N * 128;
    float*       h_nx = h0 + (long)(i + 1) * N * 128;
    hipMemsetAsync(aggr, 0, (size_t)N * 128 * sizeof(float), stream);
    edge_aggregate_kernel<<<4096, blk, 0, stream>>>(erow, ecol, eval, h_in, aggr, E);
    DenseArgs a = {};
    a.seg_ptr[0] = aggr; a.seg_k[0] = 128; a.seg_ld[0] = 128;
    a.seg_ptr[1] = h_t;  a.seg_k[1] = 128; a.seg_ld[1] = 0;   // broadcast row
    a.nseg = 2; a.M = N; a.Ktot = 256;
    a.Wt = wtG + (long)i * 128 * 256; a.ldwt = 256;
    a.bias = gnnB + i * 128; a.out = tmp; a.ldo = 128; a.relu = 1;
    dense_wmma_kernel<<<grid1, blk, 0, stream>>>(a);
    layernorm_kernel<<<2048, blk, 0, stream>>>(tmp, lnG + i * 128, lnB + i * 128,
                                               h_nx, N);
  }

  // 4) output MLP layer 1: tmp = relu([h0|h1|h2|h3|h_t] @ Wo1 + bo1)  [N,640]
  {
    DenseArgs a = {};
    for (int s = 0; s < 4; ++s) {
      a.seg_ptr[s] = h0 + (long)s * N * 128; a.seg_k[s] = 128; a.seg_ld[s] = 128;
    }
    a.seg_ptr[4] = h_t; a.seg_k[4] = 128; a.seg_ld[4] = 0;   // broadcast row
    a.nseg = 5; a.M = N; a.Ktot = 640;
    a.Wt = wtO1; a.ldwt = 640; a.bias = bo1; a.out = tmp; a.ldo = 640; a.relu = 1;
    dense_wmma_kernel<<<dim3((N + BLK_ROWS - 1) / BLK_ROWS, 5), blk, 0, stream>>>(a);
  }
  // 5) output MLP layer 2: out = tmp @ Wo2 + bo2  [N,128]
  {
    DenseArgs a = make_dense1(tmp, 640, 640, N, wtO2, bo2, out, 128, 0);
    dense_wmma_kernel<<<grid1, blk, 0, stream>>>(a);
  }
}
